// CrossAttentionLayer_541165879462
// MI455X (gfx1250) — matrix operand order repaired and checked by reading, not compiled, not run on hardware
//
#include <hip/hip_runtime.h>
#include <cstdint>

// ---------------------------------------------------------------------------
// CDNA5 / gfx1250 implementation of the E(n)-equivariant cross-attention layer.
// Matmuls run on v_wmma_f32_16x16x32_f16 (wave32, 16-edge M-tiles).
// ---------------------------------------------------------------------------

typedef __attribute__((ext_vector_type(16))) _Float16 v16h;
typedef __attribute__((ext_vector_type(8)))  float    v8f;

__device__ __forceinline__ v8f wzero() {
  v8f z;
  for (int i = 0; i < 8; ++i) z[i] = 0.f;
  return z;
}

__device__ __forceinline__ v8f wmma16(v16h a, v16h b, v8f c) {
  // D = A(16x32 f16) * B(32x16 f16) + C(16x16 f32)
  return __builtin_amdgcn_wmma_f32_16x16x32_f16(false, a, false, b, (short)0, c,
                                                false, false);
}

// B fragments live in LDS pre-packed per (frag, lane): 16 contiguous halves.
__device__ __forceinline__ v16h load_frag_lds(const _Float16* base, int frag, int lane) {
  return *(const v16h*)(base + frag * 512 + lane * 16);
}

// A-matrix 16x32 f16 lane layout (ISA 7.12.2):
//   lanes 0-15 : elements 0..7 -> K=kb..kb+7, elements 8..15 -> K=kb+16..kb+23, kb=0
//   lanes 16-31: same with kb=8
__device__ __forceinline__ v16h a_from_f32(const float* rowp, int kbase) {
  v16h a;
#pragma unroll
  for (int i = 0; i < 8; ++i) a[i] = (_Float16)rowp[kbase + i];
#pragma unroll
  for (int i = 0; i < 8; ++i) a[8 + i] = (_Float16)rowp[kbase + 16 + i];
  return a;
}

__device__ __forceinline__ v16h a_from_f16(const _Float16* rowp, int kbase) {
  v16h a;
#pragma unroll
  for (int i = 0; i < 8; ++i) a[i] = rowp[kbase + i];
#pragma unroll
  for (int i = 0; i < 8; ++i) a[8 + i] = rowp[kbase + 16 + i];
  return a;
}

// monotone float<->uint key for atomic max over floats (handles negatives/-inf)
__device__ __forceinline__ unsigned fkey(float f) {
  unsigned b = __float_as_uint(f);
  return b ^ ((unsigned)((int)b >> 31) | 0x80000000u);
}
__device__ __forceinline__ float funkey(unsigned k) {
  unsigned b = (k & 0x80000000u) ? (k ^ 0x80000000u) : ~k;
  return __uint_as_float(b);
}

// ------------------------------- small kernels -----------------------------

__global__ void copy_f32(const float* __restrict__ src, float* __restrict__ dst, long n) {
  long i = (long)blockIdx.x * blockDim.x + threadIdx.x;
  if (i < n) dst[i] = src[i];
}

__global__ void fill_u32(unsigned* __restrict__ dst, unsigned v, long n) {
  long i = (long)blockIdx.x * blockDim.x + threadIdx.x;
  if (i < n) dst[i] = v;
}

__global__ void fill_f32(float* __restrict__ dst, float v, long n) {
  long i = (long)blockIdx.x * blockDim.x + threadIdx.x;
  if (i < n) dst[i] = v;
}

// Edge geometry: coord_diff gram matrix + global sum of radial^2 (axis=0!)
__global__ void k1_radial(const float* __restrict__ coord, const int* __restrict__ row,
                          const int* __restrict__ col, float* __restrict__ ws_rad,
                          float* __restrict__ acc, int E) {
  long e = (long)blockIdx.x * blockDim.x + threadIdx.x;
  const bool act = e < E;
  const long ec = act ? e : (long)E - 1;
  const int r = row[ec], c = col[ec];
  float d[12];
#pragma unroll
  for (int i = 0; i < 12; ++i) d[i] = coord[(long)r * 12 + i] - coord[(long)c * 12 + i];
  float rad[16];
#pragma unroll
  for (int a = 0; a < 4; ++a)
#pragma unroll
    for (int b = 0; b < 4; ++b)
      rad[a * 4 + b] = d[a * 3] * d[b * 3] + d[a * 3 + 1] * d[b * 3 + 1] +
                       d[a * 3 + 2] * d[b * 3 + 2];
  if (act) {
#pragma unroll
    for (int i = 0; i < 16; ++i) ws_rad[e * 16 + i] = rad[i];
  }
  const int lane = threadIdx.x & 31;
#pragma unroll
  for (int i = 0; i < 16; ++i) {
    float s = act ? rad[i] * rad[i] : 0.f;
    for (int off = 16; off > 0; off >>= 1) s += __shfl_down(s, off, 32);
    if (lane == 0) atomicAdd(acc + i, s);
  }
}

__global__ void k2_norm(float* __restrict__ accbuf) {
  int i = threadIdx.x;
  if (i < 16) {
    float nrm = fmaxf(sqrtf(accbuf[i]), 1e-12f);
    accbuf[16 + i] = 1.0f / nrm;
  }
}

__global__ void k4_segmax(const float* __restrict__ ws_alpha, const int* __restrict__ row,
                          unsigned* __restrict__ segmax, int E) {
  long e = (long)blockIdx.x * blockDim.x + threadIdx.x;
  if (e >= E) return;
  atomicMax(segmax + row[e], fkey(ws_alpha[e]));
}

__global__ void k5_expsum(float* __restrict__ ws_alpha, const int* __restrict__ row,
                          const unsigned* __restrict__ segmax, float* __restrict__ segsum,
                          int E) {
  long e = (long)blockIdx.x * blockDim.x + threadIdx.x;
  if (e >= E) return;
  const int r = row[e];
  const float ex = __expf(ws_alpha[e] - funkey(segmax[r]));
  ws_alpha[e] = ex;  // overwrite alpha with exp
  atomicAdd(segsum + r, ex);
}

// One 32-lane wave per edge: scatter-add att*v into h_out, att*mlp*diff into coord_out
__global__ void __launch_bounds__(256)
k6_aggregate(const int* __restrict__ row, const int* __restrict__ col,
             const float* __restrict__ coord, const _Float16* __restrict__ ws_v,
             const float* __restrict__ ws_cmlp, const float* __restrict__ ws_ex,
             const float* __restrict__ segsum, float* __restrict__ out_h,
             float* __restrict__ out_coord, float* __restrict__ out_att, int E) {
  const int lane = threadIdx.x & 31;
  long e = (long)blockIdx.x * 8 + (threadIdx.x >> 5);
  if (e >= E) return;
  const int r = row[e], c = col[e];
  const float att = ws_ex[e] / segsum[r];
  if (lane == 0) out_att[e] = att;
  const float v0 = (float)ws_v[e * 64 + lane];
  const float v1 = (float)ws_v[e * 64 + 32 + lane];
  atomicAdd(out_h + (long)r * 64 + lane, att * v0);
  atomicAdd(out_h + (long)r * 64 + 32 + lane, att * v1);
  if (lane < 12) {
    const float diff = coord[(long)r * 12 + lane] - coord[(long)c * 12 + lane];
    const float cm = ws_cmlp[e * 4 + (lane / 3)];
    atomicAdd(out_coord + (long)r * 12 + lane, att * cm * diff);
  }
}

// ------------------------------- WMMA kernel -------------------------------
// 8 waves/block, 16 edges per wave-tile. Weights staged once per block into
// LDS as f16 B-fragments (Wkv de-interleaved into k/v halves; W2 N-padded).
// Frag map: [0..7]=Wq(kc0..1 x nt0..3)  [8..19]=Wk(kc0..2 x nt0..3)
//           [20..31]=Wv  [32..39]=W1  [40..41]=W2(kc0..1, nt0)
__global__ void __launch_bounds__(256)
edge_gemm_kernel(const float* __restrict__ h, const int* __restrict__ row,
                 const int* __restrict__ col, const float* __restrict__ edge_attr,
                 const float* __restrict__ Wq, const float* __restrict__ bq,
                 const float* __restrict__ Wkv, const float* __restrict__ bkv,
                 const float* __restrict__ W1, const float* __restrict__ b1,
                 const float* __restrict__ W2, const float* __restrict__ ws_rad,
                 const float* __restrict__ invn, float* __restrict__ ws_alpha,
                 _Float16* __restrict__ ws_v, float* __restrict__ ws_cmlp, int E) {
  __shared__ __align__(32) _Float16 wfrag[42 * 512];
  __shared__ __align__(32) _Float16 vtrans[8 * 1024];

  // ---- stage all weights as f16 B-fragments ----
  // B-matrix 32x16 f16 lane layout: lane n=l&15; lanes 0-15 hold K=kc*32+{0..7,16..23},
  // lanes 16-31 hold K=kc*32+{8..15,24..31} (same order as the A fragments).
  for (int t = threadIdx.x; t < 42 * 512; t += 256) {
    const int f = t >> 9;
    const int w = t & 511;
    const int l = w >> 4;
    const int i = w & 15;
    const int n = l & 15;
    const int hi = l >> 4;
    const int kloc = hi * 8 + (i & 7) + ((i >> 3) << 4);
    float val;
    if (f < 8) {
      const int kc = f >> 2, nt = f & 3;
      val = Wq[(kc * 32 + kloc) * 64 + nt * 16 + n];
    } else if (f < 20) {
      const int g = f - 8, kc = g >> 2, nt = g & 3;
      val = Wkv[(kc * 32 + kloc) * 128 + 2 * (nt * 16 + n)];      // k = even cols
    } else if (f < 32) {
      const int g = f - 20, kc = g >> 2, nt = g & 3;
      val = Wkv[(kc * 32 + kloc) * 128 + 2 * (nt * 16 + n) + 1];  // v = odd cols
    } else if (f < 40) {
      const int g = f - 32, kc = g >> 2, nt = g & 3;
      val = W1[(kc * 32 + kloc) * 64 + nt * 16 + n];
    } else {
      const int kc = f - 40;
      val = (n < 4) ? W2[(kc * 32 + kloc) * 4 + n] : 0.0f;
    }
    wfrag[t] = (_Float16)val;
  }
  __syncthreads();

  const int lane = threadIdx.x & 31;
  const int wave = threadIdx.x >> 5;
  const int m = lane & 15;   // edge-within-tile for A, N-col for C/D
  const int hi = lane >> 4;
  const long tileBase = (long)blockIdx.x * 128 + (long)wave * 16;
  if (tileBase >= E) return;

  const long eM = tileBase + m;
  const long eC = (eM < E) ? eM : (long)E - 1;
  const int ri = row[eC], ci = col[eC];
  const float* hrow = h + (long)ri * 64;
  const float* hcol = h + (long)ci * 64;
  const float* rad_e = ws_rad + eC * 16;
  const float* ea_e = edge_attr + eC * 16;

  // ---- q = h[row] @ Wq ----
  const v16h aq0 = a_from_f32(hrow, hi * 8);
  const v16h aq1 = a_from_f32(hrow, 32 + hi * 8);
  v8f qa[4];
#pragma unroll
  for (int nt = 0; nt < 4; ++nt) {
    v8f acc = wzero();
    acc = wmma16(aq0, load_frag_lds(wfrag, 0 + nt, lane), acc);
    acc = wmma16(aq1, load_frag_lds(wfrag, 4 + nt, lane), acc);
    qa[nt] = acc;
  }

  // ---- target_feat = [radial*invnorm | h[col] | edge_attr], K = 96 ----
  v16h tfa[3];
#pragma unroll
  for (int kc = 0; kc < 3; ++kc) {
    v16h a;
    const int kb = kc * 32 + hi * 8;
#pragma unroll
    for (int i = 0; i < 16; ++i) {
      const int K = kb + (i & 7) + ((i >> 3) << 4);
      float v;
      if (K < 16)      v = rad_e[K] * invn[K];
      else if (K < 80) v = hcol[K - 16];
      else             v = ea_e[K - 80];
      a[i] = (_Float16)v;
    }
    tfa[kc] = a;
  }

  // ---- k = tf @ Wk ----
  v8f ka[4];
#pragma unroll
  for (int nt = 0; nt < 4; ++nt) {
    v8f acc = wzero();
#pragma unroll
    for (int kc = 0; kc < 3; ++kc)
      acc = wmma16(tfa[kc], load_frag_lds(wfrag, 8 + kc * 4 + nt, lane), acc);
    ka[nt] = acc;
  }

  // ---- alpha = sum((q+bq)*(k+bk)) over 64 features ----
  float part[8];
#pragma unroll
  for (int r = 0; r < 8; ++r) part[r] = 0.f;
#pragma unroll
  for (int nt = 0; nt < 4; ++nt) {
    const int n = nt * 16 + m;
    const float bqv = bq[n];
    const float bkv_ = bkv[2 * n];
#pragma unroll
    for (int r = 0; r < 8; ++r) part[r] += (qa[nt][r] + bqv) * (ka[nt][r] + bkv_);
  }
#pragma unroll
  for (int off = 1; off < 16; off <<= 1) {
#pragma unroll
    for (int r = 0; r < 8; ++r) part[r] += __shfl_xor(part[r], off, 32);
  }
  if (m == 0) {  // lanes 0 (edges 0..7) and 16 (edges 8..15)
#pragma unroll
    for (int r = 0; r < 8; ++r) {
      const long eo = tileBase + hi * 8 + r;
      if (eo < E) ws_alpha[eo] = part[r];
    }
  }

  // ---- v = tf @ Wv + bv ; transpose via per-wave LDS ----
  _Float16* vb = vtrans + wave * 1024;
#pragma unroll
  for (int nt = 0; nt < 4; ++nt) {
    v8f acc = wzero();
#pragma unroll
    for (int kc = 0; kc < 3; ++kc)
      acc = wmma16(tfa[kc], load_frag_lds(wfrag, 20 + kc * 4 + nt, lane), acc);
    const float bvv = bkv[2 * (nt * 16 + m) + 1];
#pragma unroll
    for (int r = 0; r < 8; ++r)
      vb[(r + hi * 8) * 64 + nt * 16 + m] = (_Float16)(acc[r] + bvv);
  }

  // spill v tile to global (f16, row-major per edge == LDS layout)
  {
    long nEdges = E - tileBase;
    if (nEdges > 16) nEdges = 16;
    const int ndw = (int)nEdges * 32;
    const uint32_t* vb32 = (const uint32_t*)vb;
    uint32_t* dst = (uint32_t*)ws_v + tileBase * 32;
    for (int idx = lane; idx < ndw; idx += 32) dst[idx] = vb32[idx];
  }

  // ---- u = silu(v @ W1 + b1) ----
  const v16h ua0 = a_from_f16(vb + m * 64, hi * 8);
  const v16h ua1 = a_from_f16(vb + m * 64, 32 + hi * 8);
  v8f ug[4];
#pragma unroll
  for (int nt = 0; nt < 4; ++nt) {
    v8f acc = wzero();
    acc = wmma16(ua0, load_frag_lds(wfrag, 32 + nt, lane), acc);
    acc = wmma16(ua1, load_frag_lds(wfrag, 36 + nt, lane), acc);
    const float b1v = b1[nt * 16 + m];
#pragma unroll
    for (int r = 0; r < 8; ++r) {
      const float x = acc[r] + b1v;
      acc[r] = x * (1.f / (1.f + __expf(-x)));
    }
    ug[nt] = acc;
  }
  // transpose u through the same per-wave LDS region (in-order per wave)
#pragma unroll
  for (int nt = 0; nt < 4; ++nt) {
#pragma unroll
    for (int r = 0; r < 8; ++r)
      vb[(r + hi * 8) * 64 + nt * 16 + m] = (_Float16)ug[nt][r];
  }

  // ---- coord mlp = u @ W2 (N padded to 16, cols >=4 are zero) ----
  const v16h xa0 = a_from_f16(vb + m * 64, hi * 8);
  const v16h xa1 = a_from_f16(vb + m * 64, 32 + hi * 8);
  v8f cacc = wzero();
  cacc = wmma16(xa0, load_frag_lds(wfrag, 40, lane), cacc);
  cacc = wmma16(xa1, load_frag_lds(wfrag, 41, lane), cacc);
  if (m < 4) {
#pragma unroll
    for (int r = 0; r < 8; ++r) {
      const long eo = tileBase + hi * 8 + r;
      if (eo < E) ws_cmlp[eo * 4 + m] = cacc[r];
    }
  }
}

// ------------------------------- launcher ----------------------------------

extern "C" void kernel_launch(void* const* d_in, const int* in_sizes, int n_in,
                              void* d_out, int out_size, void* d_ws, size_t ws_size,
                              hipStream_t stream) {
  const float* h = (const float*)d_in[0];
  const float* coord = (const float*)d_in[1];
  const int* row = (const int*)d_in[2];
  const int* col = (const int*)d_in[3];
  const float* edge_attr = (const float*)d_in[4];
  const float* Wq = (const float*)d_in[5];
  const float* bq = (const float*)d_in[6];
  const float* Wkv = (const float*)d_in[7];
  const float* bkv = (const float*)d_in[8];
  const float* W1 = (const float*)d_in[9];
  const float* b1 = (const float*)d_in[10];
  const float* W2 = (const float*)d_in[11];

  const int N = in_sizes[0] / 64;
  const int E = in_sizes[2];

  float* out_h = (float*)d_out;                    // N*64
  float* out_coord = out_h + (size_t)N * 64;       // N*12
  float* out_att = out_coord + (size_t)N * 12;     // E

  // workspace layout
  float* ws_rad = (float*)d_ws;                              // E*16 f32
  float* ws_alpha = ws_rad + (size_t)E * 16;                 // E f32 (alpha -> exp)
  _Float16* ws_v = (_Float16*)(ws_alpha + (size_t)E);        // E*64 f16
  float* ws_cmlp = (float*)(ws_v + (size_t)E * 64);          // E*4 f32
  unsigned* segmax = (unsigned*)(ws_cmlp + (size_t)E * 4);   // N u32 keys
  float* segsum = (float*)(segmax + (size_t)N);              // N f32
  float* accbuf = segsum + (size_t)N;                        // 16 acc + 16 invnorm

  // init outputs / accumulators
  copy_f32<<<(int)(((long)N * 64 + 255) / 256), 256, 0, stream>>>(h, out_h, (long)N * 64);
  copy_f32<<<(int)(((long)N * 12 + 255) / 256), 256, 0, stream>>>(coord, out_coord,
                                                                  (long)N * 12);
  fill_u32<<<(N + 255) / 256, 256, 0, stream>>>(segmax, 0x007FFFFFu, N);  // key(-inf)
  fill_f32<<<(N + 255) / 256, 256, 0, stream>>>(segsum, 0.f, N);
  fill_f32<<<1, 32, 0, stream>>>(accbuf, 0.f, 16);

  k1_radial<<<(E + 255) / 256, 256, 0, stream>>>(coord, row, col, ws_rad, accbuf, E);
  k2_norm<<<1, 32, 0, stream>>>(accbuf);

  edge_gemm_kernel<<<(E + 127) / 128, 256, 0, stream>>>(
      h, row, col, edge_attr, Wq, bq, Wkv, bkv, W1, b1, W2, ws_rad, accbuf + 16,
      ws_alpha, ws_v, ws_cmlp, E);

  k4_segmax<<<(E + 255) / 256, 256, 0, stream>>>(ws_alpha, row, segmax, E);
  k5_expsum<<<(E + 255) / 256, 256, 0, stream>>>(ws_alpha, row, segmax, segsum, E);
  k6_aggregate<<<(E + 7) / 8, 256, 0, stream>>>(row, col, coord, ws_v, ws_cmlp, ws_alpha,
                                                segsum, out_h, out_coord, out_att, E);
}
